// CausalSelfAttention_43671227466252
// MI455X (gfx1250) — matrix operand order repaired and checked by reading, not compiled, not run on hardware
//
#include <hip/hip_runtime.h>

#define BB 4
#define TT 1024
#define HD 1024
#define NH 16
#define DH 64
#define DS 64

typedef __attribute__((ext_vector_type(16))) _Float16 v16h;
typedef __attribute__((ext_vector_type(8)))  _Float16 h8;
typedef __attribute__((ext_vector_type(8)))  float    v8f;
typedef __attribute__((ext_vector_type(4)))  unsigned int u32x4;
typedef __attribute__((ext_vector_type(8)))  int      i32x8;
typedef __attribute__((ext_vector_type(4)))  int      i32x4;

#define WMMA_F16(a,b,c) __builtin_amdgcn_wmma_f32_16x16x32_f16(false,(a),false,(b),(short)0,(c),false,false)

#if (__has_builtin(__builtin_amdgcn_tensor_load_to_lds) && __has_builtin(__builtin_amdgcn_s_wait_tensorcnt))
#define TDM_OK 1
#else
#define TDM_OK 0
#endif

// ---- fragment helpers (contiguous-k layouts, wave32) ----
// A 16x32 f16: lane row = lane&15; K elems = 16*hi + e (16 contiguous)
__device__ __forceinline__ v16h a_frag16(const _Float16* p, int hi) {
  h8 lo = *(const h8*)(p + 16 * hi);
  h8 hb = *(const h8*)(p + 16 * hi + 8);
  return __builtin_shufflevector(lo, hb, 0,1,2,3,4,5,6,7,8,9,10,11,12,13,14,15);
}
// B 32x16 f16: lane col = lane&15; K elems = 16*hi + e (16 contiguous)
__device__ __forceinline__ v16h b_frag16(const _Float16* p) {
  return *(const v16h*)p;     // caller passes p already offset by 16*hi
}

#if TDM_OK
// 2D f16 tile -> LDS via Tensor Data Mover (D# per CDNA5 ISA ch.8).
// tile_w elements per row, tile_h rows, row stride in elements. Rows packed in LDS.
__device__ __forceinline__ void tdm_load_2d(unsigned lds_off, const _Float16* g,
                                            int tile_w, int tile_h, int stride_elems) {
  unsigned long long ga = (unsigned long long)(const void*)g;
  u32x4 g0;
  g0[0] = 1u;                                       // count=1, user descriptor
  g0[1] = lds_off;                                  // LDS byte address
  g0[2] = (unsigned)ga;                             // global addr [31:0]
  g0[3] = (unsigned)((ga >> 32) & 0x01FFFFFFu) | (2u << 30);  // addr [56:32] | type=2
  i32x8 g1;
  g1[0] = 0x00010000;                               // data_size=1 (2 bytes), no mask/pad
  g1[1] = 0;                                        // atomic_addr=0 | tensor_dim0 lo16 (dim0=1<<20)
  g1[2] = 0x10;                                     // tensor_dim0 hi16 | tensor_dim1 lo16 (dim1=1<<20)
  g1[3] = (int)(0x10u | ((unsigned)tile_w << 16));  // tensor_dim1 hi16 | tile_dim0
  g1[4] = (int)((unsigned)tile_h & 0xFFFFu);        // tile_dim1 | tile_dim2=0
  g1[5] = stride_elems;                             // tensor_dim0_stride lo32
  g1[6] = 0;
  g1[7] = 0;
  i32x4 z4 = {0, 0, 0, 0};
  i32x8 z8 = {0, 0, 0, 0, 0, 0, 0, 0};
  __builtin_amdgcn_tensor_load_to_lds(g0, g1, z4, z4, z8, 0);
}
#define LDSOFF(p) ((unsigned)(unsigned long long)(const void*)(p))
#endif

// ---- generic f16 NT GEMM core: acc[4] += A[16 x K] * W[n0..n0+63, K]^T ----
__device__ __forceinline__ void gemm_core(const _Float16* __restrict__ A,
                                          const _Float16* __restrict__ W,
                                          int K, int m0, int n0, int lr, int hi,
                                          v8f acc[4]) {
  const _Float16* arow = A + (size_t)(m0 + lr) * K;
  const _Float16* w0 = W + (size_t)(n0 + lr) * K + 16 * hi;
  for (int kb = 0; kb < K; kb += 32) {
    v16h a  = a_frag16(arow + kb, hi);
    v16h b0 = b_frag16(w0 + kb);
    v16h b1 = b_frag16(w0 + (size_t)16 * K + kb);
    v16h b2 = b_frag16(w0 + (size_t)32 * K + kb);
    v16h b3 = b_frag16(w0 + (size_t)48 * K + kb);
    acc[0] = WMMA_F16(a, b0, acc[0]);
    acc[1] = WMMA_F16(a, b1, acc[1]);
    acc[2] = WMMA_F16(a, b2, acc[2]);
    acc[3] = WMMA_F16(a, b3, acc[3]);
  }
}

// ---- conversion / prep kernels ----
__global__ __launch_bounds__(256) void cvt16(const float* __restrict__ in,
                                             _Float16* __restrict__ out, int n) {
  int i = (blockIdx.x * 256 + threadIdx.x) * 8;
  if (i + 8 <= n) {
    #pragma unroll
    for (int e = 0; e < 8; ++e) out[i + e] = (_Float16)in[i + e];
  }
}

// W{q,k,v}_s[h,s,d] -> transposed f16 Wt[h,d,s]
__global__ __launch_bounds__(256) void prep_ws(const float* __restrict__ Wq,
                                               const float* __restrict__ Wk,
                                               const float* __restrict__ Wv,
                                               _Float16* __restrict__ Wqt,
                                               _Float16* __restrict__ Wkt,
                                               _Float16* __restrict__ Wvt) {
  int i = blockIdx.x * 256 + threadIdx.x;           // 3*16*64*64
  int which = i / (NH * 64 * 64);
  int rem = i % (NH * 64 * 64);
  int h = rem >> 12, d = (rem >> 6) & 63, s = rem & 63;
  const float* src = which == 0 ? Wq : which == 1 ? Wk : Wv;
  _Float16* dst = which == 0 ? Wqt : which == 1 ? Wkt : Wvt;
  dst[((size_t)h * 64 + d) * 64 + s] = (_Float16)src[((size_t)h * 64 + s) * 64 + d];
}

// ---- QKV projection: data16[4096,1024] x qkv_w16[3072,1024]^T + b ->
//      head-split f16: Qc/Kc [b,h,t,64], Vct [b,h,64,t]
__global__ __launch_bounds__(256) void gemm_qkv(
    const _Float16* __restrict__ A, const _Float16* __restrict__ W,
    const float* __restrict__ bias,
    _Float16* __restrict__ Qc, _Float16* __restrict__ Kc, _Float16* __restrict__ Vct) {
  int wave = (blockIdx.x * blockDim.x + threadIdx.x) >> 5;
  int lane = threadIdx.x & 31;
  int lr = lane & 15, hi = lane >> 4;
  int m0 = (wave / 48) << 4;
  int n0 = (wave % 48) << 6;
  v8f acc[4] = {};
  gemm_core(A, W, HD, m0, n0, lr, hi, acc);
  int part = n0 >> 10;                 // 0=Q 1=K 2=V
  int h = (n0 & 1023) >> 6;
  #pragma unroll
  for (int tn = 0; tn < 4; ++tn) {
    float bv = bias[n0 + tn * 16 + lr];
    int col = tn * 16 + lr;
    #pragma unroll
    for (int r = 0; r < 8; ++r) {
      int rowm = m0 + r + 8 * hi;
      int b = rowm >> 10, t = rowm & 1023;
      _Float16 v = (_Float16)(acc[tn][r] + bv);
      size_t bh = (size_t)b * NH + h;
      if (part == 0)      Qc[(bh * TT + t) * 64 + col] = v;
      else if (part == 1) Kc[(bh * TT + t) * 64 + col] = v;
      else                Vct[(bh * 64 + col) * TT + t] = v;
    }
  }
}

// ---- scratch projections: sin16[b,h,t,64] x Wt[h,64(d),64(s)] ->
//      Qs/Ks [b,h,t,64], Vst [b,h,64,t]
__global__ __launch_bounds__(256) void scratch_gemm(
    const _Float16* __restrict__ S,
    const _Float16* __restrict__ Wqt, const _Float16* __restrict__ Wkt,
    const _Float16* __restrict__ Wvt,
    _Float16* __restrict__ Qs, _Float16* __restrict__ Ks, _Float16* __restrict__ Vst) {
  int wave = (blockIdx.x * blockDim.x + threadIdx.x) >> 5;
  int lane = threadIdx.x & 31;
  int lr = lane & 15, hi = lane >> 4;
  const int per = BB * NH * (TT / 16);       // 4096
  int which = wave / per;
  int rem = wave % per;
  int bh = rem / (TT / 16);
  int tt = rem % (TT / 16);
  int h = bh & (NH - 1);
  const _Float16* Wt = (which == 0 ? Wqt : which == 1 ? Wkt : Wvt) + (size_t)h * 64 * 64;
  const _Float16* arow = S + (size_t)(bh * TT + tt * 16 + lr) * DS;
  v8f acc[4] = {};
  #pragma unroll
  for (int kb = 0; kb < DS; kb += 32) {
    v16h a = a_frag16(arow + kb, hi);
    #pragma unroll
    for (int tn = 0; tn < 4; ++tn) {
      // B[k=s][n=d]: Wt row d = tn*16+lr, contiguous in s
      v16h bf = b_frag16(Wt + (size_t)(tn * 16 + lr) * 64 + kb + 16 * hi);
      acc[tn] = WMMA_F16(a, bf, acc[tn]);
    }
  }
  #pragma unroll
  for (int tn = 0; tn < 4; ++tn) {
    int col = tn * 16 + lr;
    #pragma unroll
    for (int r = 0; r < 8; ++r) {
      int t = tt * 16 + r + 8 * hi;
      _Float16 v = (_Float16)acc[tn][r];
      if (which == 0)      Qs[((size_t)bh * TT + t) * 64 + col] = v;
      else if (which == 1) Ks[((size_t)bh * TT + t) * 64 + col] = v;
      else                 Vst[((size_t)bh * 64 + col) * TT + t] = v;
    }
  }
}

// ---- fused flash attention: grid = B*NH*(T/128), block = 256 (8 waves) ----
__global__ __launch_bounds__(256) void attn_kernel(
    const _Float16* __restrict__ Qc, const _Float16* __restrict__ Kc,
    const _Float16* __restrict__ Vct,
    const _Float16* __restrict__ Qs, const _Float16* __restrict__ Ks,
    const _Float16* __restrict__ Vst,
    const float* __restrict__ lam_p, const float* __restrict__ phv,
    _Float16* __restrict__ X16, float* __restrict__ sout,
    float* __restrict__ accs) {
  __shared__ _Float16 sKc[2][64 * 64];     // staged K-content tile (64 rows x 64 d)
  __shared__ _Float16 sKs[2][64 * 64];     // staged K-scratch tile
  __shared__ _Float16 pbuf[8][16 * 32];    // per-wave P staging
  int wid = threadIdx.x >> 5;
  int lane = threadIdx.x & 31;
  int lr = lane & 15, hi = lane >> 4;
  int qb = blockIdx.x & 7;
  int bh = blockIdx.x >> 3;
  int h = bh & (NH - 1);
  int q0 = qb * 128 + wid * 16;

  float lam = lam_p[0];
  float gate = (lam > 1e-8f) ? 1.0f : 0.0f;
  float c1 = gate * lam * __expf(phv[h]);
  const float invs = 0.125f;               // 1/sqrt(64)

  const _Float16* KcH = Kc + (size_t)bh * TT * 64;
  const _Float16* KsH = Ks + (size_t)bh * TT * 64;
  const _Float16* VcH = Vct + (size_t)bh * 64 * TT;  // [64 d][1024 t]
  const _Float16* VsH = Vst + (size_t)bh * 64 * TT;

  // Q fragments held in registers
  v16h aqc[2], aqs[2];
  {
    const _Float16* qrow  = Qc + ((size_t)bh * TT + q0 + lr) * 64;
    const _Float16* qsrow = Qs + ((size_t)bh * TT + q0 + lr) * 64;
    #pragma unroll
    for (int c = 0; c < 2; ++c) {
      aqc[c] = a_frag16(qrow + c * 32, hi);
      aqs[c] = a_frag16(qsrow + c * 32, hi);
    }
  }

  v8f oc[4] = {}, osv[4] = {};
  float m[8], lsum[8], rs_c[8], rq_c[8], rs_s[8], rq_s[8];
  #pragma unroll
  for (int r = 0; r < 8; ++r) {
    m[r] = -3.0e38f; lsum[r] = 0.f;
    rs_c[r] = 0.f; rq_c[r] = 0.f; rs_s[r] = 0.f; rq_s[r] = 0.f;
  }
  float d2 = 0.f;

#if TDM_OK
  if (wid == 0) {
    tdm_load_2d(LDSOFF(&sKc[0][0]), KcH, 64, 64, 64);
    tdm_load_2d(LDSOFF(&sKs[0][0]), KsH, 64, 64, 64);
  }
#endif

  for (int s = 0; s < TT / 64; ++s) {
    int buf;
#if TDM_OK
    buf = s & 1;
    if (wid == 0) {
      if (s + 1 < TT / 64) {
        tdm_load_2d(LDSOFF(&sKc[(s + 1) & 1][0]), KcH + (size_t)(s + 1) * 64 * 64, 64, 64, 64);
        tdm_load_2d(LDSOFF(&sKs[(s + 1) & 1][0]), KsH + (size_t)(s + 1) * 64 * 64, 64, 64, 64);
        __builtin_amdgcn_s_wait_tensorcnt(2);
      } else {
        __builtin_amdgcn_s_wait_tensorcnt(0);
      }
    }
    __syncthreads();
#else
    buf = 0;
    __syncthreads();
    {
      int tc = threadIdx.x;                 // 256 threads x 16 halves = 4096
      *(v16h*)&sKc[0][tc * 16] = *(const v16h*)(KcH + (size_t)s * 4096 + tc * 16);
      *(v16h*)&sKs[0][tc * 16] = *(const v16h*)(KsH + (size_t)s * 4096 + tc * 16);
    }
    __syncthreads();
#endif

    #pragma unroll
    for (int sub = 0; sub < 2; ++sub) {
      int kb = s * 64 + sub * 32;
      int kloc = sub * 32;
      // ---- S tiles (16q x 32k), content + scratch: 8 WMMAs ----
      v8f sc0 = {}, sc1 = {}, ss0 = {}, ss1 = {};
      #pragma unroll
      for (int c = 0; c < 2; ++c) {
        v16h b0  = b_frag16(&sKc[buf][(kloc + lr) * 64 + c * 32 + 16 * hi]);
        v16h b1  = b_frag16(&sKc[buf][(kloc + 16 + lr) * 64 + c * 32 + 16 * hi]);
        v16h bs0 = b_frag16(&sKs[buf][(kloc + lr) * 64 + c * 32 + 16 * hi]);
        v16h bs1 = b_frag16(&sKs[buf][(kloc + 16 + lr) * 64 + c * 32 + 16 * hi]);
        sc0 = WMMA_F16(aqc[c], b0, sc0);
        sc1 = WMMA_F16(aqc[c], b1, sc1);
        ss0 = WMMA_F16(aqs[c], bs0, ss0);
        ss1 = WMMA_F16(aqs[c], bs1, ss1);
      }
      // ---- unmasked statistics + gated logits ----
      float u0[8], u1[8];
      #pragma unroll
      for (int r = 0; r < 8; ++r) {
        float lc0 = sc0[r] * invs, lc1 = sc1[r] * invs;
        float ls0 = ss0[r] * invs, ls1 = ss1[r] * invs;
        float dd0 = c1 * ls0, dd1 = c1 * ls1;
        rs_c[r] += lc0 + lc1;  rq_c[r] += lc0 * lc0 + lc1 * lc1;
        rs_s[r] += ls0 + ls1;  rq_s[r] += ls0 * ls0 + ls1 * ls1;
        d2 += dd0 * dd0 + dd1 * dd1;
        u0[r] = lc0 + dd0; u1[r] = lc1 + dd1;
      }
      // ---- causal online softmax + P@V (skip fully-masked blocks) ----
      if (kb < q0 + 16) {
        float sc_r[8];
        #pragma unroll
        for (int r = 0; r < 8; ++r) {
          int rowq = q0 + r + 8 * hi;
          float mu0 = (kb + lr      <= rowq) ? u0[r] : -3.0e38f;
          float mu1 = (kb + 16 + lr <= rowq) ? u1[r] : -3.0e38f;
          float tm = fmaxf(mu0, mu1);
          #pragma unroll
          for (int off = 1; off < 16; off <<= 1) tm = fmaxf(tm, __shfl_xor(tm, off));
          float nm = fmaxf(m[r], tm);
          float sc = __expf(m[r] - nm);
          float p0 = __expf(mu0 - nm);
          float p1 = __expf(mu1 - nm);
          float rsum = p0 + p1;
          #pragma unroll
          for (int off = 1; off < 16; off <<= 1) rsum += __shfl_xor(rsum, off);
          lsum[r] = lsum[r] * sc + rsum;
          m[r] = nm;
          sc_r[r] = sc;
          pbuf[wid][(r + 8 * hi) * 32 + lr]      = (_Float16)p0;
          pbuf[wid][(r + 8 * hi) * 32 + 16 + lr] = (_Float16)p1;
        }
        #pragma unroll
        for (int tn = 0; tn < 4; ++tn)
          #pragma unroll
          for (int r = 0; r < 8; ++r) { oc[tn][r] *= sc_r[r]; osv[tn][r] *= sc_r[r]; }
        asm volatile("" ::: "memory");   // LDS writes precede same-wave re-read (DS in-order)
        v16h ap = a_frag16(&pbuf[wid][lr * 32], hi);
        #pragma unroll
        for (int tn = 0; tn < 4; ++tn) {
          // B[k][n=d]: transposed-V row d = tn*16+lr, contiguous along t
          v16h bc = b_frag16(VcH + (size_t)(tn * 16 + lr) * TT + kb + 16 * hi);
          v16h bs = b_frag16(VsH + (size_t)(tn * 16 + lr) * TT + kb + 16 * hi);
          oc[tn]  = WMMA_F16(ap, bc, oc[tn]);
          osv[tn] = WMMA_F16(ap, bs, osv[tn]);
        }
      }
    }
#if TDM_OK
    __syncthreads();                     // protect buffers before next TDM overwrite
#endif
  }

  // ---- outputs ----
  int b = bh >> 4;
  #pragma unroll
  for (int r = 0; r < 8; ++r) {
    float inv_l = 1.0f / lsum[r];
    int rowq = q0 + r + 8 * hi;
    #pragma unroll
    for (int tn = 0; tn < 4; ++tn) {
      int col = tn * 16 + lr;
      X16[((size_t)(b * TT + rowq)) * HD + h * DH + col] = (_Float16)(oc[tn][r] * inv_l);
      sout[((size_t)bh * TT + rowq) * DS + col]          = osv[tn][r] * inv_l;
    }
  }
  // ---- row-std ratio (ddof=1) and delta-RMS partials ----
  float ratio_sum = 0.f;
  const float invT1 = 1.0f / (TT - 1);
  #pragma unroll
  for (int r = 0; r < 8; ++r) {
    float sc = rs_c[r], qc = rq_c[r], ss = rs_s[r], qs = rq_s[r];
    #pragma unroll
    for (int off = 1; off < 16; off <<= 1) {
      sc += __shfl_xor(sc, off); qc += __shfl_xor(qc, off);
      ss += __shfl_xor(ss, off); qs += __shfl_xor(qs, off);
    }
    float varc = (qc - sc * sc * (1.0f / TT)) * invT1;
    float vars = (qs - ss * ss * (1.0f / TT)) * invT1;
    ratio_sum += sqrtf(fmaxf(vars, 0.f)) / (sqrtf(fmaxf(varc, 0.f)) + 1e-6f);
  }
  if (lr == 0) atomicAdd(&accs[1], ratio_sum);
  #pragma unroll
  for (int off = 1; off < 32; off <<= 1) d2 += __shfl_xor(d2, off);
  if (lane == 0) atomicAdd(&accs[0], d2);
}

// ---- output projection: X16[4096,1024] x out_w16[1024,1024]^T + b -> f32 ----
__global__ __launch_bounds__(256) void gemm_out(
    const _Float16* __restrict__ A, const _Float16* __restrict__ W,
    const float* __restrict__ bias, float* __restrict__ C) {
  int wave = (blockIdx.x * blockDim.x + threadIdx.x) >> 5;
  int lane = threadIdx.x & 31;
  int lr = lane & 15, hi = lane >> 4;
  int m0 = (wave / 16) << 4;
  int n0 = (wave % 16) << 6;
  v8f acc[4] = {};
  gemm_core(A, W, HD, m0, n0, lr, hi, acc);
  #pragma unroll
  for (int tn = 0; tn < 4; ++tn) {
    float bv = bias[n0 + tn * 16 + lr];
    #pragma unroll
    for (int r = 0; r < 8; ++r)
      C[(size_t)(m0 + r + 8 * hi) * HD + n0 + tn * 16 + lr] = acc[tn][r] + bv;
  }
}

__global__ void init_accs(float* accs) {
  if (threadIdx.x < 2) accs[threadIdx.x] = 0.f;
}
__global__ void finalize_stats(const float* __restrict__ accs, float* __restrict__ o) {
  o[0] = sqrtf(accs[0] / (float)((size_t)BB * NH * TT * TT));
  o[1] = accs[1] / (float)(BB * NH * TT);
}

extern "C" void kernel_launch(void* const* d_in, const int* in_sizes, int n_in,
                              void* d_out, int out_size, void* d_ws, size_t ws_size,
                              hipStream_t stream) {
  (void)in_sizes; (void)n_in; (void)out_size; (void)ws_size;
  const float* data  = (const float*)d_in[0];
  const float* sin   = (const float*)d_in[1];
  const float* qkv_w = (const float*)d_in[2];
  const float* qkv_b = (const float*)d_in[3];
  const float* out_w = (const float*)d_in[4];
  const float* out_b = (const float*)d_in[5];
  const float* Wq_s  = (const float*)d_in[6];
  const float* Wk_s  = (const float*)d_in[7];
  const float* Wv_s  = (const float*)d_in[8];
  const float* lam   = (const float*)d_in[9];
  const float* phv   = (const float*)d_in[10];
  float* out = (float*)d_out;

  float* accs = (float*)d_ws;                        // 2 floats (pad to 16)
  _Float16* hb = (_Float16*)((float*)d_ws + 16);
  const size_t SZ_DATA = (size_t)BB * TT * HD;       // 4194304
  const size_t SZ_QKVW = (size_t)3 * HD * HD;        // 3145728
  const size_t SZ_OUTW = (size_t)HD * HD;            // 1048576
  const size_t SZ_HEAD = (size_t)BB * NH * TT * 64;  // 4194304
  const size_t SZ_WS   = (size_t)NH * 64 * 64;       // 65536
  _Float16* data16 = hb;               hb += SZ_DATA;
  _Float16* qkvw16 = hb;               hb += SZ_QKVW;
  _Float16* outw16 = hb;               hb += SZ_OUTW;
  _Float16* sin16  = hb;               hb += SZ_HEAD;
  _Float16* Wqt    = hb;               hb += SZ_WS;
  _Float16* Wkt    = hb;               hb += SZ_WS;
  _Float16* Wvt    = hb;               hb += SZ_WS;
  _Float16* Qc     = hb;               hb += SZ_HEAD;
  _Float16* Kc     = hb;               hb += SZ_HEAD;
  _Float16* Vct    = hb;               hb += SZ_HEAD;
  _Float16* Qs     = hb;               hb += SZ_HEAD;
  _Float16* Ks     = hb;               hb += SZ_HEAD;
  _Float16* Vst    = hb;               hb += SZ_HEAD;
  _Float16* X16    = hb;               hb += SZ_HEAD;

  init_accs<<<1, 32, 0, stream>>>(accs);
  cvt16<<<(int)(SZ_DATA / 8 / 256), 256, 0, stream>>>(data, data16, (int)SZ_DATA);
  cvt16<<<(int)(SZ_QKVW / 8 / 256), 256, 0, stream>>>(qkv_w, qkvw16, (int)SZ_QKVW);
  cvt16<<<(int)(SZ_OUTW / 8 / 256), 256, 0, stream>>>(out_w, outw16, (int)SZ_OUTW);
  cvt16<<<(int)(SZ_HEAD / 8 / 256), 256, 0, stream>>>(sin, sin16, (int)SZ_HEAD);
  prep_ws<<<(int)(3 * SZ_WS / 256), 256, 0, stream>>>(Wq_s, Wk_s, Wv_s, Wqt, Wkt, Wvt);

  // QKV projection: (4096/16)*(3072/64) = 12288 waves -> 1536 blocks
  gemm_qkv<<<1536, 256, 0, stream>>>(data16, qkvw16, qkv_b, Qc, Kc, Vct);
  // Scratch projections: 3*4096 waves -> 1536 blocks
  scratch_gemm<<<1536, 256, 0, stream>>>(sin16, Wqt, Wkt, Wvt, Qs, Ks, Vst);
  // Fused attention: B*NH*(T/128) = 512 blocks
  attn_kernel<<<512, 256, 0, stream>>>(Qc, Kc, Vct, Qs, Ks, Vst, lam, phv,
                                       X16, out + (size_t)BB * TT * HD, accs);
  // Output projection: (4096/16)*(1024/64) = 4096 waves -> 512 blocks
  gemm_out<<<512, 256, 0, stream>>>(X16, outw16, out_b, out);
  finalize_stats<<<1, 1, 0, stream>>>(accs, out + (size_t)2 * BB * TT * HD);
}
